// LayoutGraphAwareModel_65094524338524
// MI455X (gfx1250) — compile-verified, matrix operand order checked by reading, not run on hardware
//
#include <hip/hip_runtime.h>
#include <hip/hip_bf16.h>
#include <math.h>

// ---------------------------------------------------------------------------
// Model dims (match reference setup_inputs)
// ---------------------------------------------------------------------------
#define NN    20000      // nodes
#define EE    160000     // edges
#define CC    2000       // config nodes
#define BB    8          // batch
#define HH    128        // hidden
#define LL    3          // layers
#define SIN_D 209        // 121 + 32 + 8 + 6*8
#define CIN_D 144        // 18*8
#define NNUM_D 121
#define OPD_D 32
#define TD_D   8
#define SLOTS_D 6
#define CFD_D 18
#define LVD_D  8

typedef __attribute__((ext_vector_type(16))) __bf16 v16bf;
typedef __attribute__((ext_vector_type(8)))  __bf16 v8bf;
typedef __attribute__((ext_vector_type(8)))  float  v8f;

__device__ __forceinline__ float gelu_exact(float x) {
    return 0.5f * x * (1.0f + erff(x * 0.70710678118654752f));
}
__device__ __forceinline__ __bf16 f2bf(float f) { return (__bf16)f; }

union frag_u {
    v16bf v;
    struct { v8bf lo, hi; } h;
};

// ---------------------------------------------------------------------------
// Pack fp32 weights into bf16 *fragment order*:
//   dst[((l*32 + kt*8 + nt)*32 + lane)*16 + e] = W[l][k][n]
//   k = kt*32 + (e>>3)*16 + (lane>>4)*8 + (e&7),  n = (lane&15) + nt*16
// so a B-fragment is two contiguous 16B vector loads per lane.
// ---------------------------------------------------------------------------
__global__ void k_pack_weights(const float* W, __bf16* WF, int total) {
    int d = blockIdx.x * blockDim.x + threadIdx.x;
    if (d >= total) return;
    int l = d / (HH * HH);
    int r = d % (HH * HH);
    int frag = r >> 9;          // 512 elems per fragment
    int rem  = r & 511;
    int lane = rem >> 4, e = rem & 15;
    int kt = frag >> 3, nt = frag & 7;
    int k = kt * 32 + ((e >> 3) * 16) + ((lane >> 4) * 8) + (e & 7);
    int n = (lane & 15) + nt * 16;
    WF[d] = f2bf(W[(size_t)l * HH * HH + (size_t)k * HH + n]);
}

// ---------------------------------------------------------------------------
// static_h = gelu(static_in @ Ws + bs)        one block per node, 128 threads
// ---------------------------------------------------------------------------
__global__ void k_static(const float* nnf, const int* opc, const int* nst,
                         const int* nlf, const float* op_emb, const float* sh_emb,
                         const float* lv_emb, const float* Ws, const float* bs,
                         float* staticH) {
    __shared__ float sbuf[SIN_D];
    int n = blockIdx.x, t = threadIdx.x;
    for (int k = t; k < NNUM_D; k += 128) sbuf[k] = nnf[(size_t)n * NNUM_D + k];
    if (t < OPD_D)  sbuf[NNUM_D + t] = op_emb[opc[n] * OPD_D + t];
    if (t < TD_D)   sbuf[NNUM_D + OPD_D + t] = sh_emb[nst[n] * TD_D + t];
    if (t < SLOTS_D * LVD_D) {
        int slot = t / LVD_D, d = t % LVD_D;
        sbuf[NNUM_D + OPD_D + TD_D + t] = lv_emb[nlf[n * SLOTS_D + slot] * LVD_D + d];
    }
    __syncthreads();
    float acc = bs[t];
    #pragma unroll 4
    for (int k = 0; k < SIN_D; ++k) acc += sbuf[k] * Ws[k * HH + t];
    staticH[(size_t)n * HH + t] = gelu_exact(acc);
}

// ---------------------------------------------------------------------------
// cfg_h = gelu(cfg_e @ Wc1 + bc1) @ Wc2 + bc2   one block per (b,c)
// ---------------------------------------------------------------------------
__global__ void k_cfg(const int* cfg_batch, const float* lv_emb,
                      const float* Wc1, const float* bc1,
                      const float* Wc2, const float* bc2, float* cfgH) {
    __shared__ float ce[CIN_D];
    __shared__ float hid[HH];
    int bc = blockIdx.x, t = threadIdx.x;
    for (int k = t; k < CIN_D; k += 128) {
        int slot = k / LVD_D, d = k % LVD_D;
        ce[k] = lv_emb[cfg_batch[(size_t)bc * CFD_D + slot] * LVD_D + d];
    }
    __syncthreads();
    float acc = bc1[t];
    #pragma unroll 4
    for (int k = 0; k < CIN_D; ++k) acc += ce[k] * Wc1[k * HH + t];
    hid[t] = gelu_exact(acc);
    __syncthreads();
    float acc2 = bc2[t];
    #pragma unroll 4
    for (int k = 0; k < HH; ++k) acc2 += hid[k] * Wc2[k * HH + t];
    cfgH[(size_t)bc * HH + t] = acc2;
}

// broadcast static_h over batch
__global__ void k_bcast(const float* staticH, float* hF, size_t per_batch, size_t total) {
    size_t idx = (size_t)blockIdx.x * blockDim.x + threadIdx.x;
    if (idx < total) hF[idx] = staticH[idx % per_batch];
}

// scatter-add cfg_h at node_config_ids
__global__ void k_addcfg(const float* cfgH, const int* ids, float* hF) {
    int bc = blockIdx.x, t = threadIdx.x;
    int b = bc / CC, c = bc % CC;
    int node = ids[c];
    atomicAdd(&hF[((size_t)b * NN + node) * HH + t], cfgH[(size_t)bc * HH + t]);
}

// row layernorm (128 threads per row)
__global__ void k_ln(float* hF, const float* g, const float* bta) {
    __shared__ float red[HH];
    size_t row = blockIdx.x;
    int t = threadIdx.x;
    float v = hF[row * HH + t];
    red[t] = v; __syncthreads();
    for (int s = 64; s > 0; s >>= 1) { if (t < s) red[t] += red[t + s]; __syncthreads(); }
    float mean = red[0] * (1.0f / HH); __syncthreads();
    float d = v - mean;
    red[t] = d * d; __syncthreads();
    for (int s = 64; s > 0; s >>= 1) { if (t < s) red[t] += red[t + s]; __syncthreads(); }
    float inv = rsqrtf(red[0] * (1.0f / HH) + 1e-5f);
    hF[row * HH + t] = d * inv * g[t] + bta[t];
}

__global__ void k_zero(float* p, size_t total) {
    size_t idx = (size_t)blockIdx.x * blockDim.x + threadIdx.x;
    if (idx < total) p[idx] = 0.0f;
}

// edge scatter: agg[b,dst] += h[b,src]   one block per edge, 128 threads
__global__ void k_scatter(const int* ei, const float* hF, float* aggF) {
    int e = blockIdx.x, t = threadIdx.x;
    int src = ei[e], dst = ei[EE + e];
    #pragma unroll
    for (int b = 0; b < BB; ++b) {
        float v = hF[((size_t)b * NN + src) * HH + t];
        atomicAdd(&aggF[((size_t)b * NN + dst) * HH + t], v);
    }
}

// agg /= max(deg,1) -> bf16 ; h -> bf16
__global__ void k_scalecast(const float* hF, const float* aggF, const float* deg,
                            __bf16* hB, __bf16* aggB, size_t total) {
    size_t idx = (size_t)blockIdx.x * blockDim.x + threadIdx.x;
    if (idx >= total) return;
    size_t n = (idx / HH) % NN;
    float d = fmaxf(deg[n], 1.0f);
    aggB[idx] = f2bf(aggF[idx] / d);
    hB[idx]   = f2bf(hF[idx]);
}

// ---------------------------------------------------------------------------
// WMMA fragment loaders (bf16, 16x16x32, wave32 layouts per CDNA5 ISA 7.12.2)
// ---------------------------------------------------------------------------
__device__ __forceinline__ v16bf load_a_frag(const __bf16* X, size_t row0, int kt, int lane) {
    int r  = lane & 15;
    int kh = (lane >> 4) * 8;
    const __bf16* p = X + (row0 + r) * HH + kt * 32 + kh;
    frag_u u;
    u.h.lo = *(const v8bf*)p;          // K = kt*32 + kh + 0..7
    u.h.hi = *(const v8bf*)(p + 16);   // K = kt*32 + kh + 16..23
    return u.v;
}

// fragment-ordered weights: 32B contiguous per lane
__device__ __forceinline__ v16bf load_b_frag(const __bf16* WF, int kt, int nt, int lane) {
    const __bf16* p = WF + (((size_t)(kt * 8 + nt) * 32 + lane) << 4);
    frag_u u;
    u.h.lo = *(const v8bf*)p;
    u.h.hi = *(const v8bf*)(p + 8);
    return u.v;
}

// ---------------------------------------------------------------------------
// Fused layer: z = gelu(h@Wself + bself + agg@Wneigh + bneigh); h = LN(h+z)
// 128 threads = 4 waves; wave owns 16 rows x full 128 cols.
// Pre-LN values staged in per-wave LDS tile (controlled spill); LN stats
// accumulated on the fly + cross-lane __shfl_xor reduction.
// __launch_bounds__(128,1): allow full VGPR budget -> no scratch spills
// (LDS already caps residency; spilling would be strictly worse).
// ---------------------------------------------------------------------------
__global__ void __launch_bounds__(128, 1)
k_gemm_layer(float* hF, const __bf16* hB, const __bf16* aggB,
             const __bf16* WsF, const __bf16* WnF,
             const float* bself, const float* bneigh,
             const float* g, const float* bta) {
    __shared__ float zbuf[4][16][HH];   // 32 KB
    int lane = threadIdx.x & 31;
    int w    = threadIdx.x >> 5;
    size_t row0 = (size_t)blockIdx.x * 64 + (size_t)w * 16;

    v16bf ah[4], ag[4];
    #pragma unroll
    for (int kt = 0; kt < 4; ++kt) {
        ah[kt] = load_a_frag(hB,   row0, kt, lane);
        ag[kt] = load_a_frag(aggB, row0, kt, lane);
    }

    int hh = lane >> 4;
    int n0 = lane & 15;
    float s[8], sq[8];
    #pragma unroll
    for (int i = 0; i < 8; ++i) { s[i] = 0.0f; sq[i] = 0.0f; }

    #pragma unroll
    for (int nt = 0; nt < 8; ++nt) {
        v8f c = {};
        #pragma unroll
        for (int kt = 0; kt < 4; ++kt) {
            v16bf bS = load_b_frag(WsF, kt, nt, lane);
            c = __builtin_amdgcn_wmma_f32_16x16x32_bf16(false, ah[kt], false, bS,
                                                        (short)0, c, false, false);
            v16bf bN = load_b_frag(WnF, kt, nt, lane);
            c = __builtin_amdgcn_wmma_f32_16x16x32_bf16(false, ag[kt], false, bN,
                                                        (short)0, c, false, false);
        }
        int n = nt * 16 + n0;
        float bb = bself[n] + bneigh[n];
        #pragma unroll
        for (int i = 0; i < 8; ++i) {
            float z = gelu_exact(c[i] + bb);
            float v = hF[(row0 + i + 8 * hh) * HH + n] + z;   // residual
            zbuf[w][i + 8 * hh][n] = v;
            s[i]  += v;
            sq[i] += v * v;
        }
    }

    // layernorm per row; stats reduced across the 16 lanes of this half-wave
    #pragma unroll
    for (int i = 0; i < 8; ++i) {
        float rs = s[i], rq = sq[i];
        #pragma unroll
        for (int m = 1; m < 16; m <<= 1) {
            rs += __shfl_xor(rs, m, 32);
            rq += __shfl_xor(rq, m, 32);
        }
        float mean = rs * (1.0f / HH);
        float var  = rq * (1.0f / HH) - mean * mean;
        float inv  = rsqrtf(var + 1e-5f);
        size_t row = row0 + i + 8 * hh;
        #pragma unroll
        for (int nt = 0; nt < 8; ++nt) {
            int n = nt * 16 + n0;
            float v = zbuf[w][i + 8 * hh][n];
            hF[row * HH + n] = (v - mean) * inv * g[n] + bta[n];
        }
    }
}

// ---------------------------------------------------------------------------
// attention logits s[b,c] = hc[b,c,:]·Wa + ba
// ---------------------------------------------------------------------------
__global__ void k_attn_logits(const float* hF, const int* ids, const float* Wa,
                              const float* ba, float* s) {
    __shared__ float red[HH];
    int bc = blockIdx.x, t = threadIdx.x;
    int b = bc / CC, c = bc % CC;
    int node = ids[c];
    red[t] = hF[((size_t)b * NN + node) * HH + t] * Wa[t];
    __syncthreads();
    for (int st = 64; st > 0; st >>= 1) { if (t < st) red[t] += red[t + st]; __syncthreads(); }
    if (t == 0) s[bc] = red[0] + ba[0];
}

// softmax over configs + attn/mean pooling -> head_in (B x 256)
__global__ void k_softmax_pool(const float* s, const float* hF, const int* ids,
                               float* headin) {
    __shared__ float red[256];
    __shared__ float smax, ssum;
    int b = blockIdx.x, t = threadIdx.x;
    float m = -3.0e38f;
    for (int c = t; c < CC; c += 256) m = fmaxf(m, s[b * CC + c]);
    red[t] = m; __syncthreads();
    for (int st = 128; st > 0; st >>= 1) { if (t < st) red[t] = fmaxf(red[t], red[t + st]); __syncthreads(); }
    if (t == 0) smax = red[0];
    __syncthreads();
    float sum = 0.0f;
    for (int c = t; c < CC; c += 256) sum += expf(s[b * CC + c] - smax);
    red[t] = sum; __syncthreads();
    for (int st = 128; st > 0; st >>= 1) { if (t < st) red[t] += red[t + st]; __syncthreads(); }
    if (t == 0) ssum = red[0];
    __syncthreads();

    float acc = 0.0f;
    if (t < HH) {
        for (int c = 0; c < CC; ++c) {
            float a = expf(s[b * CC + c] - smax) / ssum;
            acc += a * hF[((size_t)b * NN + ids[c]) * HH + t];
        }
    } else {
        int d = t - HH;
        for (int c = 0; c < CC; ++c)
            acc += hF[((size_t)b * NN + ids[c]) * HH + d];
        acc *= (1.0f / CC);
    }
    headin[b * 2 * HH + t] = acc;
}

// final MLP head: out[b] = gelu(head_in @ Wh1 + bh1) @ Wh2 + bh2
__global__ void k_head(const float* headin, const float* Wh1, const float* bh1,
                       const float* Wh2, const float* bh2, float* out) {
    __shared__ float hid[2 * HH];
    __shared__ float red[2 * HH];
    int t = threadIdx.x;
    for (int b = 0; b < BB; ++b) {
        float acc = bh1[t];
        #pragma unroll 4
        for (int k = 0; k < 2 * HH; ++k) acc += headin[b * 2 * HH + k] * Wh1[k * 2 * HH + t];
        hid[t] = gelu_exact(acc);
        __syncthreads();
        red[t] = hid[t] * Wh2[t];
        __syncthreads();
        for (int st = 128; st > 0; st >>= 1) { if (t < st) red[t] += red[t + st]; __syncthreads(); }
        if (t == 0) out[b] = red[0] + bh2[0];
        __syncthreads();
    }
}

// ---------------------------------------------------------------------------
// launch
// ---------------------------------------------------------------------------
extern "C" void kernel_launch(void* const* d_in, const int* in_sizes, int n_in,
                              void* d_out, int out_size, void* d_ws, size_t ws_size,
                              hipStream_t stream) {
    const float* nnf     = (const float*)d_in[0];
    const int*   opc     = (const int*)  d_in[1];
    const int*   nst     = (const int*)  d_in[2];
    const int*   nlf     = (const int*)  d_in[3];
    const int*   ids     = (const int*)  d_in[4];
    const int*   ei      = (const int*)  d_in[5];
    const float* deg     = (const float*)d_in[6];
    const int*   cfgb    = (const int*)  d_in[7];
    const float* op_emb  = (const float*)d_in[8];
    const float* sh_emb  = (const float*)d_in[9];
    const float* lv_emb  = (const float*)d_in[10];
    const float* Ws      = (const float*)d_in[11];
    const float* bs      = (const float*)d_in[12];
    const float* Wc1     = (const float*)d_in[13];
    const float* bc1     = (const float*)d_in[14];
    const float* Wc2     = (const float*)d_in[15];
    const float* bc2     = (const float*)d_in[16];
    const float* g_in    = (const float*)d_in[17];
    const float* b_in    = (const float*)d_in[18];
    const float* Wself   = (const float*)d_in[19];
    const float* bself   = (const float*)d_in[20];
    const float* Wneigh  = (const float*)d_in[21];
    const float* bneigh  = (const float*)d_in[22];
    const float* gn      = (const float*)d_in[23];
    const float* bn      = (const float*)d_in[24];
    const float* Wa      = (const float*)d_in[25];
    const float* ba      = (const float*)d_in[26];
    const float* Wh1     = (const float*)d_in[27];
    const float* bh1     = (const float*)d_in[28];
    const float* Wh2     = (const float*)d_in[29];
    const float* bh2     = (const float*)d_in[30];
    float* out = (float*)d_out;

    // workspace carve-up
    char* wsp = (char*)d_ws;
    auto alloc = [&](size_t bytes) -> void* {
        void* p = (void*)wsp;
        wsp += (bytes + 255) & ~((size_t)255);
        return p;
    };
    const size_t total = (size_t)BB * NN * HH;            // 20.48M elems
    float*  hF      = (float*) alloc(total * 4);
    float*  aggF    = (float*) alloc(total * 4);
    __bf16* hB      = (__bf16*)alloc(total * 2);
    __bf16* aggB    = (__bf16*)alloc(total * 2);
    float*  staticH = (float*) alloc((size_t)NN * HH * 4);
    float*  cfgH    = (float*) alloc((size_t)BB * CC * HH * 4);
    __bf16* WselfF  = (__bf16*)alloc((size_t)LL * HH * HH * 2);
    __bf16* WneighF = (__bf16*)alloc((size_t)LL * HH * HH * 2);
    float*  slog    = (float*) alloc((size_t)BB * CC * 4);
    float*  headin  = (float*) alloc((size_t)BB * 2 * HH * 4);

    const int wtot = LL * HH * HH;
    k_pack_weights<<<(wtot + 255) / 256, 256, 0, stream>>>(Wself,  WselfF,  wtot);
    k_pack_weights<<<(wtot + 255) / 256, 256, 0, stream>>>(Wneigh, WneighF, wtot);

    k_static<<<NN, 128, 0, stream>>>(nnf, opc, nst, nlf, op_emb, sh_emb, lv_emb, Ws, bs, staticH);
    k_cfg<<<BB * CC, 128, 0, stream>>>(cfgb, lv_emb, Wc1, bc1, Wc2, bc2, cfgH);
    k_bcast<<<(unsigned)((total + 255) / 256), 256, 0, stream>>>(staticH, hF, (size_t)NN * HH, total);
    k_addcfg<<<BB * CC, 128, 0, stream>>>(cfgH, ids, hF);
    k_ln<<<BB * NN, 128, 0, stream>>>(hF, g_in, b_in);

    for (int l = 0; l < LL; ++l) {
        k_zero<<<(unsigned)((total + 255) / 256), 256, 0, stream>>>(aggF, total);
        k_scatter<<<EE, 128, 0, stream>>>(ei, hF, aggF);
        k_scalecast<<<(unsigned)((total + 255) / 256), 256, 0, stream>>>(hF, aggF, deg, hB, aggB, total);
        k_gemm_layer<<<(BB * NN) / 64, 128, 0, stream>>>(
            hF, hB, aggB,
            WselfF + (size_t)l * HH * HH, WneighF + (size_t)l * HH * HH,
            bself + l * HH, bneigh + l * HH, gn + l * HH, bn + l * HH);
    }

    k_attn_logits<<<BB * CC, 128, 0, stream>>>(hF, ids, Wa, ba, slog);
    k_softmax_pool<<<BB, 256, 0, stream>>>(slog, hF, ids, headin);
    k_head<<<1, 256, 0, stream>>>(headin, Wh1, bh1, Wh2, bh2, out);
}
